// Head_32298154066657
// MI455X (gfx1250) — matrix operand order repaired and checked by reading, not compiled, not run on hardware
//
#include <hip/hip_runtime.h>
#include <hip/hip_bf16.h>
#include <stdint.h>
#include <math.h>

// Problem sizes (fixed by the reference)
#define BATCH 8
#define TSEQ  2048
#define CIN   1024
#define DHEAD 128
#define MROWS (BATCH * TSEQ)   // 16384
#define JPART (TSEQ / 4)       // 512-wide j partition per wave

typedef __attribute__((ext_vector_type(16))) __bf16    v16bf;
typedef __attribute__((ext_vector_type(8)))  float     v8f;
typedef __attribute__((ext_vector_type(8)))  uint32_t  v8u;

union BFrag { v8u u; v16bf b; };

// float -> bf16 (round-to-nearest-even), packed helpers
__device__ __forceinline__ uint16_t f2bf(float f) {
  uint32_t u = __float_as_uint(f);
  u += 0x7FFFu + ((u >> 16) & 1u);
  return (uint16_t)(u >> 16);
}
__device__ __forceinline__ uint32_t pack2(float lo, float hi) {
  return (uint32_t)f2bf(lo) | ((uint32_t)f2bf(hi) << 16);
}

// ---------------------------------------------------------------------------
// Kernel 0: weight prep.  W[C,D] f32  ->  Wt[3][D][C] bf16 (transposed).
// ---------------------------------------------------------------------------
__global__ void __launch_bounds__(256)
prep_w_kernel(const float* __restrict__ Wk,
              const float* __restrict__ Wq,
              const float* __restrict__ Wv,
              uint16_t* __restrict__ Wt)
{
  const int pair = blockIdx.x * 256 + threadIdx.x;   // [w][d][c/2]
  const int PERW = DHEAD * CIN / 2;                  // 65536 pairs per matrix
  const int w    = pair / PERW;
  const int rem  = pair % PERW;
  const int d    = rem / (CIN / 2);
  const int c    = (rem % (CIN / 2)) * 2;
  const float* W = (w == 0) ? Wk : (w == 1) ? Wq : Wv;
  ((uint32_t*)Wt)[pair] =
      pack2(W[(size_t)c * DHEAD + d], W[(size_t)(c + 1) * DHEAD + d]);
}

// ---------------------------------------------------------------------------
// Kernel 1: K/Q/V projection.  One block per 16-row m-tile.  X slab staged
// once into LDS as bf16 (shared by all 24 n-tiles); each wave accumulates
// 3 output tiles per A-fragment load.  V is written TRANSPOSED [B][D][T].
// ---------------------------------------------------------------------------
__global__ void __launch_bounds__(256)
proj_kernel(const float* __restrict__ X,
            const uint16_t* __restrict__ Wt,   // [3][D][C] bf16
            uint16_t* __restrict__ Kb,         // [M][D] bf16
            uint16_t* __restrict__ Qb,         // [M][D] bf16
            uint16_t* __restrict__ Vt)         // [B][D][T] bf16
{
  __shared__ uint16_t Xs[16][CIN + 8];         // +16B row pad (bank spread)

  const int lane = threadIdx.x & 31;
  const int wave = threadIdx.x >> 5;
  const int mt   = blockIdx.x;                 // 0..1023
  const int half = lane >> 4;
  const int n    = lane & 15;

  // ---- stage X[16 x 1024] f32 -> bf16 in LDS ------------------------------
  const float4* X4 = (const float4*)(X + (size_t)mt * 16 * CIN);
  for (int e = threadIdx.x; e < 16 * (CIN / 4); e += 256) {
    const int row = e >> 8;                    // 256 float4 per row
    const int c4  = (e & 255) * 4;
    const float4 x = X4[(size_t)row * (CIN / 4) + (c4 >> 2)];
    uint32_t* dst = (uint32_t*)&Xs[row][c4];
    dst[0] = pack2(x.x, x.y);
    dst[1] = pack2(x.z, x.w);
  }
  __syncthreads();

  // ---- 3 n-tiles per wave, A-fragment shared across them ------------------
  v8f acc3[3] = {};
  const uint32_t* Wrow[3];
#pragma unroll
  for (int tt = 0; tt < 3; ++tt) {
    const int nt = wave * 3 + tt;              // 0..23
    const int wsel = nt / (DHEAD / 16);
    const int ct   = nt % (DHEAD / 16);
    Wrow[tt] = (const uint32_t*)(Wt + ((size_t)wsel * DHEAD + ct * 16 + n) * CIN);
  }

  for (int k0 = 0; k0 < CIN; k0 += 32) {
    BFrag a;
#pragma unroll
    for (int v = 0; v < 8; ++v) {
      const int ka = k0 + (v >> 2) * 16 + half * 8 + (v & 3) * 2;
      a.u[v] = *(const uint32_t*)&Xs[n][ka];   // ds_load, b128-mergeable
    }
#pragma unroll
    for (int tt = 0; tt < 3; ++tt) {
      BFrag bw;
#pragma unroll
      for (int v = 0; v < 8; ++v)
        bw.u[v] = Wrow[tt][(k0 >> 1) + (v >> 2) * 8 + half * 4 + (v & 3)];
      acc3[tt] = __builtin_amdgcn_wmma_f32_16x16x32_bf16(
          false, a.b, false, bw.b, (short)0, acc3[tt], false, false);
    }
  }

  // ---- store --------------------------------------------------------------
  const int gmrow0 = mt * 16 + 8 * half;       // first of this half's 8 rows
  const int bat    = gmrow0 / TSEQ;            // uniform per block
  const int trow0  = gmrow0 - bat * TSEQ;
#pragma unroll
  for (int tt = 0; tt < 3; ++tt) {
    const int nt = wave * 3 + tt;
    const int wsel = nt / (DHEAD / 16);
    const int ct   = nt % (DHEAD / 16);
    if (wsel == 2) {
      // V transposed: 8 consecutive bf16 along t -> one 16B store
      uint32_t* dst = (uint32_t*)(Vt +
          ((size_t)(bat * DHEAD + ct * 16 + n)) * TSEQ + trow0);
      dst[0] = pack2(acc3[tt][0], acc3[tt][1]);
      dst[1] = pack2(acc3[tt][2], acc3[tt][3]);
      dst[2] = pack2(acc3[tt][4], acc3[tt][5]);
      dst[3] = pack2(acc3[tt][6], acc3[tt][7]);
    } else {
      uint16_t* Ob = (wsel == 0) ? Kb : Qb;
#pragma unroll
      for (int r = 0; r < 8; ++r)
        Ob[(size_t)(mt * 16 + r + 8 * half) * DHEAD + ct * 16 + n] =
            f2bf(acc3[tt][r]);
    }
  }
}

// ---------------------------------------------------------------------------
// Kernel 2: fused attention, j-split 4 ways for occupancy.
// Block = 128 threads = 4 waves = one 16-row i-tile; wave w owns j in
// [w*512, w*512+512) with an online softmax, producing an UNNORMALIZED
// partial (acc, rowmax, rowsum).  Partials are merged across the 4 waves in
// LDS (flash-attention combine), then normalized and stored cooperatively.
// ---------------------------------------------------------------------------
__global__ void __launch_bounds__(128)
attn_kernel(const uint16_t* __restrict__ Kb,
            const uint16_t* __restrict__ Qb,
            const uint16_t* __restrict__ Vt,   // [B][D][T]
            float* __restrict__ Out)
{
  __shared__ uint16_t Plds[4][16][32];         // per-wave 16x32 bf16 P tile
  __shared__ float    AccS[4][16][132];        // partials, padded row stride
  __shared__ float    Mst[4][16], Lst[4][16];

  const int lane = threadIdx.x & 31;
  const int wave = threadIdx.x >> 5;           // j-partition 0..3
  const int bat  = blockIdx.x / (TSEQ / 16);
  const int it   = blockIdx.x % (TSEQ / 16);
  const int half = lane >> 4;
  const int n    = lane & 15;

  const float scale = 0.08838834764831845f;    // 128^-0.5
  const int   jbeg  = wave * JPART;
  const int   jend  = jbeg + JPART;

  const uint32_t* Krow =
      (const uint32_t*)(Kb + ((size_t)bat * TSEQ + it * 16 + n) * DHEAD);
  const uint32_t* VtBase = (const uint32_t*)(Vt + (size_t)bat * DHEAD * TSEQ);

  v8f   acc[8];
  float rmax[8], lsum[8];
#pragma unroll
  for (int r = 0; r < 8; ++r) { acc[r] = (v8f){}; rmax[r] = -INFINITY; lsum[r] = 0.0f; }

  for (int j0 = jbeg; j0 < jend; j0 += 32) {
    // ---- S(16x32) = K(16x128) @ Q^T(128x32) -------------------------------
    const uint32_t* Qr0 =
        (const uint32_t*)(Qb + ((size_t)bat * TSEQ + j0 + n) * DHEAD);
    const uint32_t* Qr1 =
        (const uint32_t*)(Qb + ((size_t)bat * TSEQ + j0 + 16 + n) * DHEAD);
    v8f s0 = {}, s1 = {};
#pragma unroll
    for (int s = 0; s < 4; ++s) {              // 4 k-steps over d = 128
      BFrag a, b0, b1;
#pragma unroll
      for (int v = 0; v < 8; ++v) {
        a.u[v]  = Krow[s * 16 + (v >> 2) * 8 + half * 4 + (v & 3)];
        b0.u[v] = Qr0[s * 16 + (v >> 2) * 8 + half * 4 + (v & 3)];
        b1.u[v] = Qr1[s * 16 + (v >> 2) * 8 + half * 4 + (v & 3)];
      }
      s0 = __builtin_amdgcn_wmma_f32_16x16x32_bf16(
          false, a.b, false, b0.b, (short)0, s0, false, false);
      s1 = __builtin_amdgcn_wmma_f32_16x16x32_bf16(
          false, a.b, false, b1.b, (short)0, s1, false, false);
    }

    if (j0 + 32 < jend) {                      // prefetch next j-block
      __builtin_prefetch(Qb + ((size_t)bat * TSEQ + j0 + 32 + n) * DHEAD, 0, 1);
      __builtin_prefetch(Vt + ((size_t)bat * DHEAD + n) * TSEQ + j0 + 32, 0, 1);
    }

    // ---- online softmax (acc VGPR r == matrix row r + 8*half) -------------
    float bm[8], al[8], ps[8];
#pragma unroll
    for (int r = 0; r < 8; ++r) {
      s0[r] *= scale; s1[r] *= scale;
      bm[r] = fmaxf(s0[r], s1[r]);
    }
#pragma unroll
    for (int x = 1; x < 16; x <<= 1)
#pragma unroll
      for (int r = 0; r < 8; ++r)
        bm[r] = fmaxf(bm[r], __shfl_xor(bm[r], x, 32));
#pragma unroll
    for (int r = 0; r < 8; ++r) {
      const float nm = fmaxf(rmax[r], bm[r]);
      al[r] = __expf(rmax[r] - nm);
      rmax[r] = nm;
      s0[r] = __expf(s0[r] - nm);
      s1[r] = __expf(s1[r] - nm);
      ps[r] = s0[r] + s1[r];
    }
#pragma unroll
    for (int x = 1; x < 16; x <<= 1)
#pragma unroll
      for (int r = 0; r < 8; ++r)
        ps[r] += __shfl_xor(ps[r], x, 32);
#pragma unroll
    for (int r = 0; r < 8; ++r) lsum[r] = lsum[r] * al[r] + ps[r];
#pragma unroll
    for (int t = 0; t < 8; ++t)
#pragma unroll
      for (int r = 0; r < 8; ++r) acc[t][r] *= al[r];

    // ---- P through LDS: C-layout -> bf16 A-fragment -----------------------
    __syncthreads();
#pragma unroll
    for (int r = 0; r < 8; ++r) {
      Plds[wave][r + 8 * half][n]      = f2bf(s0[r]);
      Plds[wave][r + 8 * half][16 + n] = f2bf(s1[r]);
    }
    __syncthreads();
    BFrag pa;
#pragma unroll
    for (int v = 0; v < 8; ++v) {
      const int kp = (v >> 2) * 16 + half * 8 + (v & 3) * 2;
      pa.u[v] = *(const uint32_t*)&Plds[wave][n][kp];
    }

    // ---- acc(16x128) += P(16x32) @ V(32x128), V from transposed buffer ----
#pragma unroll
    for (int t = 0; t < 8; ++t) {
      const uint32_t* VtRow = VtBase + (size_t)(t * 16 + n) * (TSEQ / 2);
      BFrag bv;
#pragma unroll
      for (int v = 0; v < 8; ++v)
        bv.u[v] = VtRow[(j0 >> 1) + (v >> 2) * 8 + half * 4 + (v & 3)];
      acc[t] = __builtin_amdgcn_wmma_f32_16x16x32_bf16(
          false, pa.b, false, bv.b, (short)0, acc[t], false, false);
    }
  }

  // ---- publish this wave's unnormalized partial to LDS --------------------
#pragma unroll
  for (int t = 0; t < 8; ++t)
#pragma unroll
    for (int r = 0; r < 8; ++r)
      AccS[wave][r + 8 * half][t * 16 + n] = acc[t][r];
  if (n == 0) {
#pragma unroll
    for (int r = 0; r < 8; ++r) {
      Mst[wave][r + 8 * half] = rmax[r];
      Lst[wave][r + 8 * half] = lsum[r];
    }
  }
  __syncthreads();

  // ---- merge 4 partials (flash combine), normalize, store -----------------
  const int gt  = threadIdx.x;                 // 0..127
  const int row = gt >> 3;                     // 16 rows x 8 threads
  const int c0  = (gt & 7) * 16;               // 16 consecutive cols each
  float m = -INFINITY;
#pragma unroll
  for (int p = 0; p < 4; ++p) m = fmaxf(m, Mst[p][row]);
  float wgt[4], l = 0.0f;
#pragma unroll
  for (int p = 0; p < 4; ++p) {
    wgt[p] = __expf(Mst[p][row] - m);
    l += wgt[p] * Lst[p][row];
  }
  const float inv = 1.0f / l;
  float* orow = Out + ((size_t)bat * TSEQ + it * 16 + row) * DHEAD + c0;
#pragma unroll
  for (int c = 0; c < 16; c += 4) {
    float4 o; o.x = o.y = o.z = o.w = 0.0f;
#pragma unroll
    for (int p = 0; p < 4; ++p) {
      o.x += wgt[p] * AccS[p][row][c0 + c + 0];
      o.y += wgt[p] * AccS[p][row][c0 + c + 1];
      o.z += wgt[p] * AccS[p][row][c0 + c + 2];
      o.w += wgt[p] * AccS[p][row][c0 + c + 3];
    }
    o.x *= inv; o.y *= inv; o.z *= inv; o.w *= inv;
    *(float4*)(orow + c) = o;
  }
}

// ---------------------------------------------------------------------------
extern "C" void kernel_launch(void* const* d_in, const int* in_sizes, int n_in,
                              void* d_out, int out_size, void* d_ws, size_t ws_size,
                              hipStream_t stream) {
  const float* X  = (const float*)d_in[0];
  const float* Wk = (const float*)d_in[1];
  const float* Wq = (const float*)d_in[2];
  const float* Wv = (const float*)d_in[3];

  // Workspace: Kb, Qb row-major [M,D]; Vt transposed [B][D][T]; Wt [3][D][C]
  uint16_t* Kb = (uint16_t*)d_ws;
  uint16_t* Qb = Kb + (size_t)MROWS * DHEAD;
  uint16_t* Vt = Qb + (size_t)MROWS * DHEAD;
  uint16_t* Wt = Vt + (size_t)MROWS * DHEAD;   // +768 KB

  // 3*128*1024/2 = 196608 bf16-pairs / 256 threads = 768 blocks
  prep_w_kernel<<<768, 256, 0, stream>>>(Wk, Wq, Wv, Wt);
  // one block per 16-row m-tile
  proj_kernel<<<1024, 256, 0, stream>>>(X, Wt, Kb, Qb, Vt);
  // one block (4 waves, 4 j-partitions) per i-tile: 1024 blocks, 4096 waves
  attn_kernel<<<1024, 128, 0, stream>>>(Kb, Qb, Vt, (float*)d_out);
}
